// MLA_19026705121897
// MI455X (gfx1250) — matrix operand order repaired and checked by reading, not compiled, not run on hardware
//
#include <hip/hip_runtime.h>
#include <hip/hip_bf16.h>
#include <math.h>

// ---------------------------------------------------------------------------
// MLA forward for MI455X (gfx1250, wave32, WMMA 16x16x32 f16,
// double-buffered async global->LDS staging).
// ---------------------------------------------------------------------------

typedef _Float16 hv8  __attribute__((ext_vector_type(8)));
typedef _Float16 hv16 __attribute__((ext_vector_type(16)));
typedef float    fv8  __attribute__((ext_vector_type(8)));

static constexpr int T_   = 2048;
static constexpr int D_   = 2048;
static constexpr int NH_  = 16;
static constexpr int DN_  = 128;
static constexpr int DR_  = 64;
static constexpr int DV_  = 128;
static constexpr int RQ_  = 1536;
static constexpr int RKV_ = 512;
static constexpr int DK_  = DN_ + DR_;         // 192
static constexpr int NQ_  = NH_ * DK_;         // 3072
static constexpr int NKV_ = NH_ * (DN_ + DV_); // 4096
static constexpr float EPS_ = 1e-6f;
static constexpr float LOG_THETA_ = 9.210340371976184f; // ln(10000)

__device__ __forceinline__ hv16 cat8(hv8 lo, hv8 hi) {
  return __builtin_shufflevector(lo, hi, 0,1,2,3,4,5,6,7,8,9,10,11,12,13,14,15);
}

// Async 16-byte global -> LDS copy (CDNA5 GLOBAL_LOAD_ASYNC_TO_LDS_B128,
// tracked by ASYNCcnt). LDS offset = low 32 bits of the flat shared address.
__device__ __forceinline__ void async_cp16(void* lds, const void* gptr) {
  unsigned ldsoff = (unsigned)(unsigned long long)lds;
  asm volatile("global_load_async_to_lds_b128 %0, %1, off"
               :: "v"(ldsoff), "v"((unsigned long long)gptr) : "memory");
}
__device__ __forceinline__ void async_wait0() {
  asm volatile("s_wait_asynccnt 0x0" ::: "memory");
}

// ---------------------------------------------------------------------------
// f16 GEMM: C[M,N] = A[M,K] * Bt[N,K]^T ; A row-major f16, Bt = B transposed
// f16 [N,K]. M%128==0, K%32==0; N guarded (Bt rows padded to 128 multiple).
// Block = 256 threads (8 waves); tile 128x128; wave = 32(M) x 64(N) -> 8 WMMA
// per k-step. Double-buffered async staging overlaps copy of tile i+1 with
// WMMA compute of tile i.
// ---------------------------------------------------------------------------
__global__ __launch_bounds__(256) void gemm_h(
    const _Float16* __restrict__ A, const _Float16* __restrict__ Bt,
    float* __restrict__ C, int M, int N, int K)
{
  __shared__ alignas(16) _Float16 Ash[2][128][40];   // ping-pong [m][k] + pad
  __shared__ alignas(16) _Float16 Bsh[2][128][40];   // ping-pong [n][k] + pad

  const int tid  = threadIdx.x;
  const int wave = tid >> 5;
  const int lane = tid & 31;
  const int g    = lane >> 4;
  const int ln   = lane & 15;
  const int mbase = (wave & 3) * 32;
  const int nbase = (wave >> 2) * 64;
  const int bm = blockIdx.y * 128;
  const int bn = blockIdx.x * 128;

  // staging assignment: 4 threads per 32-half row, 4 async b128/thread/tile
  const int r0 = tid >> 2;
  const int c0 = (tid & 3) * 8;
  const _Float16* ga0 = A  + (size_t)(bm + r0)      * K + c0;
  const _Float16* ga1 = A  + (size_t)(bm + r0 + 64) * K + c0;
  const _Float16* gb0 = Bt + (size_t)(bn + r0)      * K + c0;
  const _Float16* gb1 = Bt + (size_t)(bn + r0 + 64) * K + c0;

  fv8 acc[2][4] = {};

  // prologue: stage tile 0 into buffer 0
  async_cp16(&Ash[0][r0][c0],      ga0);
  async_cp16(&Ash[0][r0 + 64][c0], ga1);
  async_cp16(&Bsh[0][r0][c0],      gb0);
  async_cp16(&Bsh[0][r0 + 64][c0], gb1);

  const int nk = K >> 5;
  for (int i = 0; i < nk; ++i) {
    async_wait0();        // tile i copies (this wave) complete
    __syncthreads();      // all waves: copies visible, prev compute done
    const int buf = i & 1;
    if (i + 1 < nk) {     // overlap: stage tile i+1 into the other buffer
      const int nb = buf ^ 1;
      const size_t koff = (size_t)(i + 1) * 32;
      async_cp16(&Ash[nb][r0][c0],      ga0 + koff);
      async_cp16(&Ash[nb][r0 + 64][c0], ga1 + koff);
      async_cp16(&Bsh[nb][r0][c0],      gb0 + koff);
      async_cp16(&Bsh[nb][r0 + 64][c0], gb1 + koff);
    }

    hv16 af[2];
#pragma unroll
    for (int mi = 0; mi < 2; ++mi) {
      const int arow = mbase + mi * 16 + ln;
      af[mi] = cat8(*(const hv8*)&Ash[buf][arow][g * 8],
                    *(const hv8*)&Ash[buf][arow][16 + g * 8]);
    }
#pragma unroll
    for (int nt = 0; nt < 4; ++nt) {
      const int brow = nbase + nt * 16 + ln;
      hv16 bf = cat8(*(const hv8*)&Bsh[buf][brow][g * 8],
                     *(const hv8*)&Bsh[buf][brow][16 + g * 8]);
#pragma unroll
      for (int mi = 0; mi < 2; ++mi)
        acc[mi][nt] = __builtin_amdgcn_wmma_f32_16x16x32_f16(
            false, af[mi], false, bf, (short)0, acc[mi][nt], false, false);
    }
  }

#pragma unroll
  for (int mi = 0; mi < 2; ++mi) {
#pragma unroll
    for (int nt = 0; nt < 4; ++nt) {
#pragma unroll
      for (int v = 0; v < 8; ++v) {
        const int row = bm + mbase + mi * 16 + v + 8 * g;
        const int col = bn + nbase + nt * 16 + ln;
        if (col < N) C[(size_t)row * N + col] = acc[mi][nt][v];
      }
    }
  }
}

// ---------------------------------------------------------------------------
// Weight prep: f32 [K,N] -> f16 transposed [N,K] (LDS-tiled transpose).
// grid (N/32, K/32), block 256 (= 32x8).
// ---------------------------------------------------------------------------
__global__ __launch_bounds__(256) void transpose_to_f16(
    const float* __restrict__ in, _Float16* __restrict__ out, int K, int N)
{
  __shared__ float tile[32][33];
  const int tx = threadIdx.x & 31, ty = threadIdx.x >> 5;
  const int n0 = blockIdx.x * 32, k0 = blockIdx.y * 32;
#pragma unroll
  for (int j = 0; j < 32; j += 8)
    tile[ty + j][tx] = in[(size_t)(k0 + ty + j) * N + n0 + tx];
  __syncthreads();
#pragma unroll
  for (int j = 0; j < 32; j += 8)
    out[(size_t)(n0 + ty + j) * K + k0 + tx] = (_Float16)tile[tx][ty + j];
}

// Plain f32 -> f16 convert, 8 elements/thread.
__global__ __launch_bounds__(256) void cvt_f16(
    const float* __restrict__ in, _Float16* __restrict__ out, int n)
{
  const int i = (blockIdx.x * 256 + threadIdx.x) * 8;
  if (i >= n) return;
  const float4 a = *(const float4*)(in + i);
  const float4 b = *(const float4*)(in + i + 4);
  hv8 h = {(_Float16)a.x, (_Float16)a.y, (_Float16)a.z, (_Float16)a.w,
           (_Float16)b.x, (_Float16)b.y, (_Float16)b.z, (_Float16)b.w};
  *(hv8*)(out + i) = h;
}

// ---------------------------------------------------------------------------
// Row-wise RMSNorm, f32 in -> f16 out (feeds next GEMM as A operand).
// ---------------------------------------------------------------------------
__global__ __launch_bounds__(256) void rmsnorm_h(
    const float* __restrict__ in, int in_stride,
    _Float16* __restrict__ out, int out_stride,
    const float* __restrict__ gamma, int n)
{
  __shared__ float red[256];
  const int row = blockIdx.x;
  const float* ip = in + (size_t)row * in_stride;
  float ss = 0.f;
  for (int i = threadIdx.x; i < n; i += 256) { float v = ip[i]; ss += v * v; }
  red[threadIdx.x] = ss;
  __syncthreads();
  for (int s = 128; s > 0; s >>= 1) {
    if (threadIdx.x < s) red[threadIdx.x] += red[threadIdx.x + s];
    __syncthreads();
  }
  const float r = rsqrtf(red[0] / (float)n + EPS_);
  _Float16* op = out + (size_t)row * out_stride;
  for (int i = threadIdx.x; i < n; i += 256)
    op[i] = (_Float16)(ip[i] * r * gamma[i]);
}

// ---------------------------------------------------------------------------
// Pack Q heads to f16 with RoPE on last 64 dims, pre-scaled by 1/sqrt(192).
// ---------------------------------------------------------------------------
__global__ __launch_bounds__(192) void build_qh16(
    const float* __restrict__ q, _Float16* __restrict__ qh)
{
  const int t = blockIdx.x, n = blockIdx.y, i = threadIdx.x;
  const float* src = q + ((size_t)t * NH_ + n) * DK_;
  const float scale = 0.07216878364870323f;   // (DN+DR)^-0.5
  float val;
  if (i < DN_) {
    val = src[i];
  } else {
    const int j = (i - DN_) & 31;
    const float invf = expf(-(float)j * (LOG_THETA_ / 32.f));
    const float ang = (float)t * invf;
    const float c = cosf(ang), s = sinf(ang);
    const float x1 = src[DN_ + j], x2 = src[DN_ + 32 + j];
    val = (i < DN_ + 32) ? (x1 * c - x2 * s) : (x2 * c + x1 * s);
  }
  qh[((size_t)t * NH_ + n) * DK_ + i] = (_Float16)(val * scale);
}

__global__ __launch_bounds__(64) void build_kp16(
    const float* __restrict__ kv, _Float16* __restrict__ kp)
{
  const int t = blockIdx.x, i = threadIdx.x;
  const float* src = kv + (size_t)t * (RKV_ + DR_) + RKV_;
  const int j = i & 31;
  const float invf = expf(-(float)j * (LOG_THETA_ / 32.f));
  const float ang = (float)t * invf;
  const float c = cosf(ang), s = sinf(ang);
  const float x1 = src[j], x2 = src[32 + j];
  kp[(size_t)t * DR_ + i] = (_Float16)((i < 32) ? (x1 * c - x2 * s)
                                                : (x2 * c + x1 * s));
}

__global__ __launch_bounds__(256) void build_knv16(
    const float* __restrict__ kvb, _Float16* __restrict__ kn,
    _Float16* __restrict__ vv)
{
  const int t = blockIdx.x, n = blockIdx.y, i = threadIdx.x;
  const float* src = kvb + ((size_t)t * NH_ + n) * 256;
  if (i < 128) kn[((size_t)t * NH_ + n) * 128 + i] = (_Float16)src[i];
  else         vv[((size_t)t * NH_ + n) * 128 + (i - 128)] = (_Float16)src[i];
}

// ---------------------------------------------------------------------------
// Flash-style causal attention. grid (T/64, NH); block 128 = 4 waves.
// Double-buffered K/V staging: K via async b128 copies (ASYNCcnt), V staged
// transposed (hv8 loads + b16 scatter). Output f16 feeds the final GEMM.
// ---------------------------------------------------------------------------
__global__ __launch_bounds__(128) void mla_attention(
    const _Float16* __restrict__ qh, const _Float16* __restrict__ kn,
    const _Float16* __restrict__ kp, const _Float16* __restrict__ vv,
    _Float16* __restrict__ outp)
{
  __shared__ alignas(16) _Float16 Ksh[2][32][200];   // [key][192] + pad
  __shared__ alignas(16) _Float16 Vsh[2][128][40];   // [vcol][key] + pad
  __shared__ alignas(16) _Float16 Psh[4][16][40];    // per-wave P tile

  const int tid  = threadIdx.x;
  const int wave = tid >> 5;
  const int lane = tid & 31;
  const int g    = lane >> 4;
  const int ln   = lane & 15;
  const int head = blockIdx.y;
  const int qrow0 = blockIdx.x * 64 + wave * 16;

  hv16 qf[6];
  {
    const _Float16* qrow = qh + ((size_t)(qrow0 + ln) * NH_ + head) * DK_;
#pragma unroll
    for (int c = 0; c < 6; ++c)
      qf[c] = cat8(*(const hv8*)(qrow + c * 32 + g * 8),
                   *(const hv8*)(qrow + c * 32 + 16 + g * 8));
  }

  fv8 o[8] = {};
  float mx[8], ls[8];
#pragma unroll
  for (int v = 0; v < 8; ++v) { mx[v] = -INFINITY; ls[v] = 0.f; }

  const int nkb = blockIdx.x * 2 + 2;

  // stage key-block kb into buffer b
  auto stage = [&](int kb, int b) {
    const int k0 = kb * 32;
    for (int c = tid; c < 768; c += 128) {      // K: 768 async b128 chunks
      const int r = c / 24, j = c - r * 24;
      if (j < 16)
        async_cp16(&Ksh[b][r][j * 8],
                   kn + ((size_t)(k0 + r) * NH_ + head) * DN_ + j * 8);
      else
        async_cp16(&Ksh[b][r][DN_ + (j - 16) * 8],
                   kp + (size_t)(k0 + r) * DR_ + (j - 16) * 8);
    }
    for (int idx = tid; idx < 512; idx += 128) { // V: hv8 load + b16 scatter
      const int r = idx >> 4, cc = (idx & 15) * 8;
      hv8 v = *(const hv8*)(vv + ((size_t)(k0 + r) * NH_ + head) * DV_ + cc);
#pragma unroll
      for (int j = 0; j < 8; ++j) Vsh[b][cc + j][r] = v[j];
    }
  };

  stage(0, 0);
  for (int kb = 0; kb < nkb; ++kb) {
    const int k0 = kb * 32;
    async_wait0();
    __syncthreads();   // emits s_wait_dscnt before barrier: V stores visible
    const int buf = kb & 1;
    if (kb + 1 < nkb) stage(kb + 1, buf ^ 1);

    if (k0 > qrow0 + 15) continue;   // wave-uniform: fully masked block

    fv8 sc[2] = {};
#pragma unroll
    for (int s = 0; s < 2; ++s) {
#pragma unroll
      for (int c = 0; c < 6; ++c) {
        const _Float16* kr = &Ksh[buf][s * 16 + ln][c * 32 + g * 8];
        hv16 bf = cat8(*(const hv8*)kr, *(const hv8*)(kr + 16));
        sc[s] = __builtin_amdgcn_wmma_f32_16x16x32_f16(
            false, qf[c], false, bf, (short)0, sc[s], false, false);
      }
    }

#pragma unroll
    for (int v = 0; v < 8; ++v) {
      const int qg = qrow0 + v + 8 * g;
      float a = (k0 + ln      <= qg) ? sc[0][v] : -INFINITY;
      float b = (k0 + 16 + ln <= qg) ? sc[1][v] : -INFINITY;
      float rm = fmaxf(a, b);
      rm = fmaxf(rm, __shfl_xor(rm, 1));
      rm = fmaxf(rm, __shfl_xor(rm, 2));
      rm = fmaxf(rm, __shfl_xor(rm, 4));
      rm = fmaxf(rm, __shfl_xor(rm, 8));
      const float nm = fmaxf(mx[v], rm);
      const float alpha = expf(mx[v] - nm);
      const float p0 = expf(a - nm);
      const float p1 = expf(b - nm);
      float rs = p0 + p1;
      rs += __shfl_xor(rs, 1);
      rs += __shfl_xor(rs, 2);
      rs += __shfl_xor(rs, 4);
      rs += __shfl_xor(rs, 8);
      ls[v] = ls[v] * alpha + rs;
      mx[v] = nm;
#pragma unroll
      for (int t2 = 0; t2 < 8; ++t2) o[t2][v] *= alpha;
      Psh[wave][v + 8 * g][ln]      = (_Float16)p0;
      Psh[wave][v + 8 * g][16 + ln] = (_Float16)p1;
    }

    asm volatile("s_wait_dscnt 0" ::: "memory");

    hv16 pf;
    {
      const _Float16* pr = &Psh[wave][ln][0];
      pf = cat8(*(const hv8*)(pr + g * 8), *(const hv8*)(pr + 16 + g * 8));
    }
#pragma unroll
    for (int t2 = 0; t2 < 8; ++t2) {
      const _Float16* vr = &Vsh[buf][t2 * 16 + ln][g * 8];
      hv16 vf = cat8(*(const hv8*)vr, *(const hv8*)(vr + 16));
      o[t2] = __builtin_amdgcn_wmma_f32_16x16x32_f16(
          false, pf, false, vf, (short)0, o[t2], false, false);
    }
  }

#pragma unroll
  for (int t2 = 0; t2 < 8; ++t2) {
#pragma unroll
    for (int v = 0; v < 8; ++v) {
      const int row = qrow0 + v + 8 * g;
      const int col = head * DV_ + t2 * 16 + ln;
      outp[(size_t)row * (NH_ * DV_) + col] = (_Float16)(o[t2][v] / ls[v]);
    }
  }
}

// ---------------------------------------------------------------------------
// Host launcher
// ---------------------------------------------------------------------------
extern "C" void kernel_launch(void* const* d_in, const int* in_sizes, int n_in,
                              void* d_out, int out_size, void* d_ws, size_t ws_size,
                              hipStream_t stream)
{
  (void)in_sizes; (void)n_in; (void)out_size; (void)ws_size;
  const float* x         = (const float*)d_in[0];
  const float* w_q_a     = (const float*)d_in[2];
  const float* q_a_norm  = (const float*)d_in[3];
  const float* w_q_b     = (const float*)d_in[4];
  const float* w_kv_a    = (const float*)d_in[5];
  const float* kv_a_norm = (const float*)d_in[6];
  const float* w_kv_b    = (const float*)d_in[7];
  const float* w_o       = (const float*)d_in[8];
  float* out = (float*)d_out;

  char* ws = (char*)d_ws;
  size_t off = 0;
  auto alloc = [&](size_t bytes) -> void* {
    void* p = ws + off;
    off += (bytes + 255) & ~(size_t)255;
    return p;
  };

  const int NKVA  = RKV_ + DR_;           // 576
  const int NKVAp = 640;                  // padded to 128-multiple for GEMM tiles

  _Float16* xh    = (_Float16*)alloc((size_t)T_ * D_ * 2);
  _Float16* wqaT  = (_Float16*)alloc((size_t)RQ_ * D_ * 2);
  _Float16* wqbT  = (_Float16*)alloc((size_t)NQ_ * RQ_ * 2);
  _Float16* wkvaT = (_Float16*)alloc((size_t)NKVAp * D_ * 2);
  _Float16* wkvbT = (_Float16*)alloc((size_t)NKV_ * RKV_ * 2);
  _Float16* woT   = (_Float16*)alloc((size_t)D_ * (NH_ * DV_) * 2);
  float*    q_c   = (float*)alloc((size_t)T_ * RQ_ * 4);
  _Float16* q_c_h = (_Float16*)alloc((size_t)T_ * RQ_ * 2);
  float*    qbuf  = (float*)alloc((size_t)T_ * NQ_ * 4);
  float*    kv    = (float*)alloc((size_t)T_ * NKVA * 4);
  _Float16* kvc_h = (_Float16*)alloc((size_t)T_ * RKV_ * 2);
  float*    kvb   = (float*)alloc((size_t)T_ * NKV_ * 4);
  _Float16* qh16  = (_Float16*)alloc((size_t)T_ * NH_ * DK_ * 2);
  _Float16* kn16  = (_Float16*)alloc((size_t)T_ * NH_ * DN_ * 2);
  _Float16* kp16  = (_Float16*)alloc((size_t)T_ * DR_ * 2);
  _Float16* v16b  = (_Float16*)alloc((size_t)T_ * NH_ * DV_ * 2);
  _Float16* attnh = (_Float16*)alloc((size_t)T_ * NH_ * DV_ * 2);

  // Operand prep: x -> f16, weights -> f16 transposed [N,K]
  cvt_f16<<<(T_ * D_ / 8 + 255) / 256, 256, 0, stream>>>(x, xh, T_ * D_);
  transpose_to_f16<<<dim3(RQ_ / 32, D_ / 32),   256, 0, stream>>>(w_q_a,  wqaT,  D_,   RQ_);
  transpose_to_f16<<<dim3(NQ_ / 32, RQ_ / 32),  256, 0, stream>>>(w_q_b,  wqbT,  RQ_,  NQ_);
  transpose_to_f16<<<dim3(NKVA / 32, D_ / 32),  256, 0, stream>>>(w_kv_a, wkvaT, D_,   NKVA);
  transpose_to_f16<<<dim3(NKV_ / 32, RKV_ / 32),256, 0, stream>>>(w_kv_b, wkvbT, RKV_, NKV_);
  transpose_to_f16<<<dim3(D_ / 32, (NH_ * DV_) / 32), 256, 0, stream>>>(w_o, woT, NH_ * DV_, D_);

  // 1) q_c = x @ w_q_a ; rmsnorm -> f16
  gemm_h<<<dim3(RQ_ / 128, T_ / 128), 256, 0, stream>>>(xh, wqaT, q_c, T_, RQ_, D_);
  rmsnorm_h<<<T_, 256, 0, stream>>>(q_c, RQ_, q_c_h, RQ_, q_a_norm, RQ_);
  // 2) q = q_c @ w_q_b
  gemm_h<<<dim3(NQ_ / 128, T_ / 128), 256, 0, stream>>>(q_c_h, wqbT, qbuf, T_, NQ_, RQ_);
  // 3) kv = x @ w_kv_a (N=576, padded tiles) ; rmsnorm first 512 -> f16
  gemm_h<<<dim3((NKVA + 127) / 128, T_ / 128), 256, 0, stream>>>(xh, wkvaT, kv, T_, NKVA, D_);
  rmsnorm_h<<<T_, 256, 0, stream>>>(kv, NKVA, kvc_h, RKV_, kv_a_norm, RKV_);
  // 4) kvb = kv_c @ w_kv_b
  gemm_h<<<dim3(NKV_ / 128, T_ / 128), 256, 0, stream>>>(kvc_h, wkvbT, kvb, T_, NKV_, RKV_);
  // 5) f16 attention operands (RoPE fused, q pre-scaled)
  build_qh16<<<dim3(T_, NH_), 192, 0, stream>>>(qbuf, qh16);
  build_kp16<<<T_, 64, 0, stream>>>(kv, kp16);
  build_knv16<<<dim3(T_, NH_), 256, 0, stream>>>(kvb, kn16, v16b);
  // 6) causal flash attention -> f16
  mla_attention<<<dim3(T_ / 64, NH_), 128, 0, stream>>>(qh16, kn16, kp16, v16b, attnh);
  // 7) out = attn @ w_o
  gemm_h<<<dim3(D_ / 128, T_ / 128), 256, 0, stream>>>(attnh, woT, out, T_, D_, NH_ * DV_);
}
